// NonLocalLayer_20950850470188
// MI455X (gfx1250) — matrix operand order repaired and checked by reading, not compiled, not run on hardware
//
#include <hip/hip_runtime.h>
#include <hip/hip_bf16.h>
#include <math.h>

// ---------------------------------------------------------------------------
// NonLocalLayer for MI455X (gfx1250, wave32, WMMA).
//   N=8, S=256, L=4096, C_in=512, C=128.
// FLOPs ~13G; dominant cost = 512MB fp32 output write (~25us @ 23.3TB/s HBM).
// GEMMs: v_wmma_f32_16x16x32_f16 (f16 in / f32 accum).
// Async path: global_load_async_to_lds_b128 + s_wait_asynccnt for V staging.
// Final stream: nontemporal stores (output 512MB > 192MB L2, zero reuse).
// ---------------------------------------------------------------------------

typedef __attribute__((ext_vector_type(16))) _Float16 v16h;
typedef __attribute__((ext_vector_type(8)))  _Float16 h8;
typedef __attribute__((ext_vector_type(8)))  float    v8f;
typedef __attribute__((ext_vector_type(4)))  float    fvec4;

#define NB    8
#define SDIM  256
#define LDIM  4096
#define CIN   512
#define CLAT  128

__device__ __forceinline__ v16h mk16(h8 lo, h8 hi) {
  return __builtin_shufflevector(lo, hi, 0,1,2,3,4,5,6,7,8,9,10,11,12,13,14,15);
}

// A-matrix fragment (16x32 f16, MxK). Lane: m = lane&15, g = lane>>4.
// elements e<8 -> k = 8g+e ; e>=8 -> k = 16+8g+(e-8)  (two contiguous 8-half runs)
__device__ __forceinline__ v16h load_a(const _Float16* base, int stride, int m, int g) {
  const h8* p0 = (const h8*)(base + m * stride + 8 * g);
  const h8* p1 = (const h8*)(base + m * stride + 16 + 8 * g);
  return mk16(*p0, *p1);
}

// B-matrix fragment (32x16 f16, KxN) from K-major storage [n][k].
// Lane: n = lane&15, g = lane>>4; elements e<8 -> k = 8g+e ; e>=8 -> k = 16+8g+(e-8)
__device__ __forceinline__ v16h load_b(const _Float16* base, int stride, int n, int g) {
  const h8* p0 = (const h8*)(base + n * stride + 8 * g);
  const h8* p1 = (const h8*)(base + n * stride + 16 + 8 * g);
  return mk16(*p0, *p1);
}

__device__ __forceinline__ v8f wmma16(v16h a, v16h b, v8f c) {
  return __builtin_amdgcn_wmma_f32_16x16x32_f16(false, a, false, b, (short)0, c,
                                                false, false);
}

// Low 32 bits of a generic pointer to a __shared__ object == LDS byte offset
// (flat->LDS mapping truncates to addr[31:0]).
__device__ __forceinline__ unsigned lds_off(const void* p) {
  return (unsigned)(uintptr_t)p;
}

// ---------------------------------------------------------------------------
// Kernel 1: projection GEMM  Y(f16)[M x 128] = X(f32)[M x 512] @ W(f32)[512 x 128] + b
// Workgroup tile: 128 rows x 128 cols, 8 waves, each wave 16 rows x 128 cols.
// ---------------------------------------------------------------------------
__global__ __launch_bounds__(256) void proj_kernel(const float* __restrict__ X,
                                                   const float* __restrict__ W,
                                                   const float* __restrict__ bias,
                                                   _Float16* __restrict__ Y) {
  __shared__ __align__(16) _Float16 Xs[128 * 32];  // [row][k]
  __shared__ __align__(16) _Float16 Ws[128 * 32];  // [n][k] transposed weight slab

  const int tid = threadIdx.x;
  const int lane = tid & 31;
  const int wave = tid >> 5;      // 0..7
  const int ln16 = lane & 15;
  const int g = lane >> 4;
  const int mBlk = blockIdx.x * 128;

  v8f acc[8];
#pragma unroll
  for (int i = 0; i < 8; ++i) acc[i] = (v8f)0.0f;

  for (int k0 = 0; k0 < CIN; k0 += 32) {
    __syncthreads();
    // Stage X tile (128x32 f32 -> f16), 8 threads/row x float4.
    {
      const int rt = tid >> 3;
      const int ct = (tid & 7) * 4;
#pragma unroll
      for (int p = 0; p < 4; ++p) {
        const int r = rt + p * 32;
        fvec4 v = *(const fvec4*)(X + (size_t)(mBlk + r) * CIN + k0 + ct);
        _Float16* d = &Xs[r * 32 + ct];
        d[0] = (_Float16)v[0]; d[1] = (_Float16)v[1];
        d[2] = (_Float16)v[2]; d[3] = (_Float16)v[3];
      }
    }
    // Stage W slab transposed: W[k0+k][n] -> Ws[n][k]
    {
      const int n0 = (tid & 31) * 4;
#pragma unroll
      for (int p = 0; p < 4; ++p) {
        const int k = (tid >> 5) + p * 8;
        fvec4 v = *(const fvec4*)(W + (size_t)(k0 + k) * CLAT + n0);
#pragma unroll
        for (int j = 0; j < 4; ++j) Ws[(n0 + j) * 32 + k] = (_Float16)v[j];
      }
    }
    __syncthreads();

    v16h a = load_a(Xs, 32, wave * 16 + ln16, g);
#pragma unroll
    for (int nt = 0; nt < 8; ++nt) {
      v16h b = load_b(Ws, 32, nt * 16 + ln16, g);
      acc[nt] = wmma16(a, b, acc[nt]);
    }
  }

  // Epilogue: bias add, convert to f16, store row-major [M][128]
#pragma unroll
  for (int nt = 0; nt < 8; ++nt) {
    const int n = nt * 16 + ln16;
    const float bi = bias[n];
#pragma unroll
    for (int r = 0; r < 8; ++r) {
      const int m = mBlk + wave * 16 + r + 8 * g;
      Y[(size_t)m * CLAT + n] = (_Float16)(acc[nt][r] + bi);
    }
  }
}

// ---------------------------------------------------------------------------
// Kernel 2: flash attention (raw-reshape semantics).
//   theta_r = theta buffer per batch viewed (128 c x 256 s) row-major
//   phi_r,g_r =                 viewed (128 c x 4096 l) row-major
//   O[s][c] = softmax_l( theta_r^T phi_r / sqrt(128) ) @ g_r^T ; write out_r[c][s].
// Grid: (S/128, N). 8 waves; wave owns 16 s-rows; online softmax over L.
// V tiles staged via ASYNC loads to LDS (pure copy, layout already [c][l]).
// ---------------------------------------------------------------------------
__global__ __launch_bounds__(256) void attn_kernel(const _Float16* __restrict__ theta,
                                                   const _Float16* __restrict__ phi,
                                                   const _Float16* __restrict__ gbuf,
                                                   float* __restrict__ outr) {
  __shared__ __align__(16) _Float16 Qt[128 * 128];      // [s_local][c]
  __shared__ __align__(16) _Float16 KT[128 * 128];      // [l_local][c]
  __shared__ __align__(16) _Float16 Vs[128 * 128];      // [c][l_local] (async-staged)
  __shared__ __align__(16) _Float16 Pt[8 * 16 * 128];   // per-wave [s16][l128]

  const int n = blockIdx.y;
  const int sBlk = blockIdx.x * 128;
  const int tid = threadIdx.x;
  const int lane = tid & 31;
  const int wave = tid >> 5;
  const int ln16 = lane & 15;
  const int g = lane >> 4;

  const _Float16* th = theta + (size_t)n * CLAT * SDIM;
  const _Float16* ph = phi + (size_t)n * CLAT * LDIM;
  const _Float16* gv = gbuf + (size_t)n * CLAT * LDIM;
  const unsigned long long gv64 = (unsigned long long)(uintptr_t)gv;

  // Stage Q transposed: Qt[s][c] = theta_r[c][sBlk+s]
#pragma unroll
  for (int it = 0; it < 8; ++it) {
    const int c = (tid >> 4) + it * 16;
    const int s = (tid & 15) * 8;
    h8 v = *(const h8*)(th + (size_t)c * SDIM + sBlk + s);
#pragma unroll
    for (int j = 0; j < 8; ++j) Qt[(s + j) * 128 + c] = v[j];
  }

  v8f O[8];
#pragma unroll
  for (int i = 0; i < 8; ++i) O[i] = (v8f)0.0f;
  float mrow[8], lrow[8];
#pragma unroll
  for (int r = 0; r < 8; ++r) { mrow[r] = -INFINITY; lrow[r] = 0.0f; }

  const float scale = 0.08838834764831845f;  // 1/sqrt(128)
  _Float16* P = &Pt[wave * 16 * 128];

  for (int l0 = 0; l0 < LDIM; l0 += 128) {
    __syncthreads();

    // Async-stage V tile: Vs[c][l] = g_r[c][l0+l]. 2048 x 16B chunks, 8/thread.
#pragma unroll
    for (int p = 0; p < 8; ++p) {
      const int i = tid + p * 256;
      const int c = i >> 4;
      const int off = (i & 15) * 16;                       // bytes within 256B row
      const unsigned voff = (unsigned)c * (LDIM * 2) + (unsigned)l0 * 2 + off;
      const unsigned dst = lds_off(&Vs[0]) + (unsigned)c * 256 + off;
      asm volatile("global_load_async_to_lds_b128 %0, %1, %2"
                   :: "v"(dst), "v"(voff), "s"(gv64) : "memory");
    }

    // Stage K tile transposed: KT[l][c] = phi_r[c][l0+l]
#pragma unroll
    for (int it = 0; it < 8; ++it) {
      const int c = (tid >> 4) + it * 16;
      const int ll = (tid & 15) * 8;
      h8 v = *(const h8*)(ph + (size_t)c * LDIM + l0 + ll);
#pragma unroll
      for (int j = 0; j < 8; ++j) KT[(ll + j) * 128 + c] = v[j];
    }
    // Prefetch next K tile (global_prefetch_b8)
    if (l0 + 128 < LDIM)
      __builtin_prefetch(ph + (size_t)(tid >> 1) * LDIM + l0 + 128 + (tid & 1) * 64, 0, 1);

    asm volatile("s_wait_asynccnt 0x0" ::: "memory");
    __syncthreads();

    // GEMM1: S(16 x 128) = Q(16 x 128c) @ K^T
    v8f S[8];
#pragma unroll
    for (int i = 0; i < 8; ++i) S[i] = (v8f)0.0f;
#pragma unroll
    for (int c0 = 0; c0 < 128; c0 += 32) {
      v16h a = load_a(Qt + c0, 128, wave * 16 + ln16, g);
#pragma unroll
      for (int nt = 0; nt < 8; ++nt) {
        v16h b = load_b(KT + c0, 128, nt * 16 + ln16, g);
        S[nt] = wmma16(a, b, S[nt]);
      }
    }

    // Online softmax. Row m = r + 8*g lives in 16 lanes with the same g;
    // shfl_xor masks 1..8 keep the halves independent on wave32.
    float tmax[8];
#pragma unroll
    for (int r = 0; r < 8; ++r) tmax[r] = -INFINITY;
#pragma unroll
    for (int nt = 0; nt < 8; ++nt)
#pragma unroll
      for (int r = 0; r < 8; ++r) {
        S[nt][r] *= scale;
        tmax[r] = fmaxf(tmax[r], S[nt][r]);
      }
#pragma unroll
    for (int r = 0; r < 8; ++r)
      for (int msk = 1; msk < 16; msk <<= 1)
        tmax[r] = fmaxf(tmax[r], __shfl_xor(tmax[r], msk, 32));

    float corr[8], psum[8];
#pragma unroll
    for (int r = 0; r < 8; ++r) {
      float mnew = fmaxf(mrow[r], tmax[r]);
      corr[r] = __expf(mrow[r] - mnew);
      mrow[r] = mnew;
      psum[r] = 0.0f;
    }
#pragma unroll
    for (int nt = 0; nt < 8; ++nt)
#pragma unroll
      for (int r = 0; r < 8; ++r) {
        float p = __expf(S[nt][r] - mrow[r]);
        S[nt][r] = p;
        psum[r] += p;
      }
#pragma unroll
    for (int r = 0; r < 8; ++r) {
      for (int msk = 1; msk < 16; msk <<= 1)
        psum[r] += __shfl_xor(psum[r], msk, 32);
      lrow[r] = lrow[r] * corr[r] + psum[r];
    }
#pragma unroll
    for (int ct = 0; ct < 8; ++ct)
#pragma unroll
      for (int r = 0; r < 8; ++r) O[ct][r] *= corr[r];

    // P tile to LDS (f16). Same-wave LDS ops are in-order (DScnt); a compiler
    // scheduling barrier suffices before re-reading.
#pragma unroll
    for (int nt = 0; nt < 8; ++nt)
#pragma unroll
      for (int r = 0; r < 8; ++r)
        P[(r + 8 * g) * 128 + nt * 16 + ln16] = (_Float16)S[nt][r];
    __builtin_amdgcn_wave_barrier();

    // GEMM2: O(16s x 128c) += P(16 x 128l) @ V(l x c); V B-frags from LDS Vs.
#pragma unroll
    for (int ls = 0; ls < 128; ls += 32) {
      v16h a = load_a(P + ls, 128, ln16, g);
#pragma unroll
      for (int ct = 0; ct < 8; ++ct) {
        v16h b = load_b(Vs + ls, 128, ct * 16 + ln16, g);
        O[ct] = wmma16(a, b, O[ct]);
      }
    }
  }

  // Epilogue: O / rowsum, write out_r[c][s] (f32)
  float* op = outr + (size_t)n * CLAT * SDIM;
#pragma unroll
  for (int ct = 0; ct < 8; ++ct) {
    const int c = ct * 16 + ln16;
#pragma unroll
    for (int r = 0; r < 8; ++r) {
      const int s = sBlk + wave * 16 + r + 8 * g;
      op[(size_t)c * SDIM + s] = O[ct][r] / lrow[r];
    }
  }
}

// ---------------------------------------------------------------------------
// Kernel 3: per-batch LayerNorm statistics over 128*256 values.
// ---------------------------------------------------------------------------
__global__ __launch_bounds__(256) void ln_stats_kernel(const float* __restrict__ outr,
                                                       float* __restrict__ stats) {
  const int n = blockIdx.x;
  const float* x = outr + (size_t)n * CLAT * SDIM;
  float s = 0.0f, s2 = 0.0f;
  for (int i = threadIdx.x; i < CLAT * SDIM; i += 256) {
    float v = x[i];
    s += v;
    s2 += v * v;
  }
  __shared__ float rs[256], rs2[256];
  rs[threadIdx.x] = s;
  rs2[threadIdx.x] = s2;
  __syncthreads();
  for (int off = 128; off > 0; off >>= 1) {
    if (threadIdx.x < off) {
      rs[threadIdx.x] += rs[threadIdx.x + off];
      rs2[threadIdx.x] += rs2[threadIdx.x + off];
    }
    __syncthreads();
  }
  if (threadIdx.x == 0) {
    const float inv = 1.0f / (CLAT * SDIM);
    float mean = rs[0] * inv;
    float var = rs2[0] * inv - mean * mean;
    stats[2 * n] = mean;
    stats[2 * n + 1] = rsqrtf(var + 1e-3f);
  }
}

// ---------------------------------------------------------------------------
// Kernel 4: LN affine + ReLU + broadcast vs w_out/b_out. 512MB of float4
// stores -> the bandwidth-limiting kernel. Output (512MB) exceeds L2 (192MB)
// with zero reuse -> nontemporal stores (TH=NT) to avoid L2 thrash.
// Block = 256 threads = 2 (n,c,s) rows x 128 threads x float4.
// ---------------------------------------------------------------------------
__global__ __launch_bounds__(256) void final_kernel(const float* __restrict__ outr,
                                                    const float* __restrict__ stats,
                                                    const float* __restrict__ gamma,
                                                    const float* __restrict__ beta,
                                                    const float* __restrict__ w_out,
                                                    const float* __restrict__ b_out,
                                                    float* __restrict__ out) {
  const size_t row = (size_t)blockIdx.x * 2 + (threadIdx.x >> 7);  // 0..262143
  const int k = (threadIdx.x & 127) * 4;
  const int n = (int)(row >> 15);        // row / 32768
  const int cs = (int)(row & 32767);     // c*256 + s
  const float mean = stats[2 * n];
  const float rstd = stats[2 * n + 1];
  float v = (outr[row] - mean) * rstd * gamma[cs] + beta[cs];
  v = fmaxf(v, 0.0f);
  fvec4 w = *(const fvec4*)(w_out + k);
  fvec4 bo = *(const fvec4*)(b_out + k);
  fvec4 o;
#pragma unroll
  for (int i = 0; i < 4; ++i) o[i] = v * w[i] + bo[i];
  __builtin_nontemporal_store(o, (fvec4*)(out + row * (size_t)CIN + k));
}

// ---------------------------------------------------------------------------
extern "C" void kernel_launch(void* const* d_in, const int* in_sizes, int n_in,
                              void* d_out, int out_size, void* d_ws, size_t ws_size,
                              hipStream_t stream) {
  const float* st_feat = (const float*)d_in[0];   // (8,256,1,1,512)
  const float* lt_feat = (const float*)d_in[1];   // (8,4096,1,1,512)
  const float* w_st = (const float*)d_in[2];
  const float* b_st = (const float*)d_in[3];
  const float* w_lt = (const float*)d_in[4];
  const float* b_lt = (const float*)d_in[5];
  const float* w_g = (const float*)d_in[6];
  const float* b_g = (const float*)d_in[7];
  const float* ln_gamma = (const float*)d_in[8];  // (128,256,1,1)
  const float* ln_beta = (const float*)d_in[9];
  const float* w_out = (const float*)d_in[10];    // (512,)
  const float* b_out = (const float*)d_in[11];
  float* out = (float*)d_out;

  // Workspace carve (~18.3 MB, 256B-aligned chunks)
  char* ws = (char*)d_ws;
  _Float16* theta = (_Float16*)ws;               ws += (size_t)NB * SDIM * CLAT * 2;  // 512KB
  _Float16* phi   = (_Float16*)ws;               ws += (size_t)NB * LDIM * CLAT * 2;  // 8MB
  _Float16* gb    = (_Float16*)ws;               ws += (size_t)NB * LDIM * CLAT * 2;  // 8MB
  float*    outr  = (float*)ws;                  ws += (size_t)NB * CLAT * SDIM * 4;  // 1MB
  float*    stats = (float*)ws;                  ws += 256;

  // 1) Projections (f32 in -> f16 out, WMMA f16)
  proj_kernel<<<(NB * SDIM) / 128, 256, 0, stream>>>(st_feat, w_st, b_st, theta);
  proj_kernel<<<(NB * LDIM) / 128, 256, 0, stream>>>(lt_feat, w_lt, b_lt, phi);
  proj_kernel<<<(NB * LDIM) / 128, 256, 0, stream>>>(lt_feat, w_g, b_g, gb);

  // 2) Flash attention with raw-reshape views (WMMA + async LDS staging)
  attn_kernel<<<dim3(SDIM / 128, NB), 256, 0, stream>>>(theta, phi, gb, outr);

  // 3) LayerNorm stats per batch
  ln_stats_kernel<<<NB, 256, 0, stream>>>(outr, stats);

  // 4) LN + ReLU + broadcast out_conv (512MB nontemporal streaming write)
  final_kernel<<<(NB * CLAT * SDIM) / 2, 256, 0, stream>>>(outr, stats, ln_gamma,
                                                           ln_beta, w_out, b_out, out);
}
